// GATv2Encoder_62964220559635
// MI455X (gfx1250) — compile-verified, matrix operand order checked by reading, not run on hardware
//
#include <hip/hip_runtime.h>
#include <math.h>

// ---------------------------------------------------------------------------
// GATv2 encoder for MI455X (gfx1250, wave32, WMMA f32_16x16x32_f16)
// ---------------------------------------------------------------------------

typedef _Float16 f16;
typedef __attribute__((ext_vector_type(16))) _Float16 v16h;
typedef __attribute__((ext_vector_type(8)))  _Float16 v8h;
typedef __attribute__((ext_vector_type(8)))  float    v8f;

#define NN 50000          // nodes
#define EE 200000         // edges
#define DD 128            // channels
#define HH 4              // heads
#define HC 512            // H*C
#define RR 8              // relations
#define PAD_E (EE + 128)  // per-relation 16-alignment padding headroom
#define NTILES_E (PAD_E / 16)
#define MTILES_N (NN / 16)   // 50000/16 = 3125 exactly

// ---- WMMA fragment helpers -------------------------------------------------
// ISA 7.12.2, 16-bit A 16x32: lane = (m | half<<4); per lane the 16 halfs are
//   k = k0 + 8*half + [0..7]  (elements 0..7)   and
//   k = k0 + 16 + 8*half + [0..7] (elements 8..15)
// i.e. two contiguous 16B chunks of the K-major row. B assumed symmetric
// (n = lane&15, same K striping), so we pre-transpose weights to K-major.
__device__ inline v16h frag_row(const f16* __restrict__ row, int k0, int h8) {
  v8h lo = *reinterpret_cast<const v8h*>(row + k0 + h8);
  v8h hi = *reinterpret_cast<const v8h*>(row + k0 + 16 + h8);
  return __builtin_shufflevector(lo, hi, 0,1,2,3,4,5,6,7,8,9,10,11,12,13,14,15);
}
__device__ inline v16h frag_row_lds(const f16* row, int k0, int h8) {
  v8h lo = *reinterpret_cast<const v8h*>(row + k0 + h8);
  v8h hi = *reinterpret_cast<const v8h*>(row + k0 + 16 + h8);
  return __builtin_shufflevector(lo, hi, 0,1,2,3,4,5,6,7,8,9,10,11,12,13,14,15);
}
__device__ inline v8f wmma16(v16h a, v16h b, v8f c) {
  return __builtin_amdgcn_wmma_f32_16x16x32_f16(false, a, false, b, (short)0, c,
                                                false, false);
}

// order-preserving float <-> uint map (for atomic segment-max)
__device__ inline unsigned enc_f(float f) {
  unsigned u = __float_as_uint(f);
  return (u & 0x80000000u) ? ~u : (u | 0x80000000u);
}
__device__ inline float dec_f(unsigned e) {
  return (e & 0x80000000u) ? __uint_as_float(e ^ 0x80000000u)
                           : __uint_as_float(~e);
}

// ---- prep kernels ----------------------------------------------------------
__global__ void k_f32_to_f16(const float* __restrict__ s, f16* __restrict__ d, int n) {
  int i = blockIdx.x * blockDim.x + threadIdx.x;
  if (i < n) d[i] = (f16)s[i];
}

// W [batch][K][N] f32 -> WT [batch][N][K] f16 (K-major rows for B fragments)
__global__ void k_transpose(const float* __restrict__ W, f16* __restrict__ WT,
                            int K, int N, int batch) {
  int i = blockIdx.x * blockDim.x + threadIdx.x;
  int tot = batch * K * N;
  if (i >= tot) return;
  int b = i / (K * N); int rem = i - b * K * N;
  int k = rem / N;     int n = rem - k * N;
  WT[((size_t)b * N + n) * K + k] = (f16)W[i];
}

// ---- relation bucketing (counting sort into 16-aligned buckets) ------------
__global__ void k_hist(const int* __restrict__ et, int* counts, int E) {
  int i = blockIdx.x * blockDim.x + threadIdx.x;
  if (i < E) atomicAdd(&counts[et[i]], 1);
}
__global__ void k_scan(const int* __restrict__ counts, int* po, int* cursor) {
  if (threadIdx.x == 0 && blockIdx.x == 0) {
    int o = 0;
    for (int r = 0; r < RR; ++r) {
      po[r] = o; cursor[r] = o;
      o += (counts[r] + 15) & ~15;          // keep each bucket 16-aligned
    }
    po[RR] = o;
  }
}
__global__ void k_scatter(const int* __restrict__ et, int* cursor,
                          int* __restrict__ order, int E) {
  int i = blockIdx.x * blockDim.x + threadIdx.x;
  if (i < E) {
    int p = atomicAdd(&cursor[et[i]], 1);
    order[p] = i;
  }
}

// ---- node projection: x = embs @ W_l + b_l  -> [NN, 512] f32 ---------------
// one wave per 16x16 output tile; K=128 -> 4 WMMAs
__global__ __launch_bounds__(32) void k_nodeproj(const f16* __restrict__ embs16,
                                                 const f16* __restrict__ WlT,
                                                 const float* __restrict__ b_l,
                                                 float* __restrict__ x) {
  int bid = blockIdx.x;
  int mtile = bid >> 5, ntile = bid & 31;
  int lane = threadIdx.x, nloc = lane & 15, half = lane >> 4, h8 = half * 8;
  const f16* arow = embs16 + (size_t)(mtile * 16 + nloc) * DD;
  const f16* brow = WlT + (size_t)(ntile * 16 + nloc) * DD;
  v8f acc = {0, 0, 0, 0, 0, 0, 0, 0};
  for (int k0 = 0; k0 < DD; k0 += 32) {
    v16h a = frag_row(arow, k0, h8);
    v16h b = frag_row(brow, k0, h8);
    acc = wmma16(a, b, acc);
  }
  int col = ntile * 16 + nloc;
  float bv = b_l[col];
  for (int j = 0; j < 8; ++j) {
    int row = mtile * 16 + j + h8;                 // C layout: M = vgpr + 8*half
    x[(size_t)row * HC + col] = acc[j] + bv;
  }
}

// ---- fused edge kernel -----------------------------------------------------
// per 16-edge tile (single relation): e_emb @ rel[r] -> GELU -> @ W_e -> +x_i+x_j
// -> leaky_relu -> dot(att) -> logits + atomicMax per target
__global__ __launch_bounds__(32) void k_edge(
    const int* __restrict__ order, const int* __restrict__ po,
    const int* __restrict__ src, const int* __restrict__ trg,
    const f16* __restrict__ embs16, const f16* __restrict__ relT,
    const f16* __restrict__ WeT, const float* __restrict__ x,
    const float* __restrict__ att, float* __restrict__ logits,
    unsigned* __restrict__ maxenc) {
  __shared__ int s_e[16], s_src[16], s_trg[16];
  __shared__ __align__(16) f16 s_ea[16 * DD];     // edge_attr tile (GELU'd), f16

  int base = blockIdx.x * 16;
  int total = po[RR];
  if (base >= total) return;
  int r = 0;
  #pragma unroll
  for (int i = 1; i < RR; ++i) if (base >= po[i]) r = i;

  int lane = threadIdx.x, nloc = lane & 15, half = lane >> 4, h8 = half * 8;
  if (lane < 16) {
    int e = order[base + lane];
    s_e[lane] = e;
    int ec = e < 0 ? 0 : e;                        // padded slot -> any valid edge
    s_src[lane] = src[ec];
    s_trg[lane] = trg[ec];
  }
  __syncthreads();

  // --- matmul 1: [16 x 256] @ relT[r]^T -> [16 x 128] ---
  const f16* srow = embs16 + (size_t)s_src[nloc] * DD;
  const f16* trow = embs16 + (size_t)s_trg[nloc] * DD;
  const f16* bbase = relT + (size_t)r * DD * (2 * DD);
  v8f acc[8];
  #pragma unroll
  for (int nt = 0; nt < 8; ++nt) acc[nt] = (v8f){0, 0, 0, 0, 0, 0, 0, 0};
  for (int k0 = 0; k0 < 2 * DD; k0 += 32) {
    const f16* arow = (k0 < DD) ? srow : trow;     // e_emb = [embs[src]|embs[trg]]
    int kk = (k0 < DD) ? k0 : k0 - DD;
    v16h a = frag_row(arow, kk, h8);
    #pragma unroll
    for (int nt = 0; nt < 8; ++nt) {
      v16h b = frag_row(bbase + (size_t)(nt * 16 + nloc) * (2 * DD), k0, h8);
      acc[nt] = wmma16(a, b, acc[nt]);
    }
  }
  // GELU (exact erf) -> f16 tile in LDS
  #pragma unroll
  for (int nt = 0; nt < 8; ++nt)
    #pragma unroll
    for (int j = 0; j < 8; ++j) {
      float v = acc[nt][j];
      float g = 0.5f * v * (1.0f + erff(v * 0.70710678118654752f));
      s_ea[(j + h8) * DD + nt * 16 + nloc] = (f16)g;
    }
  __syncthreads();

  // --- matmul 2 + epilogue: e_p tile, z = lrelu(x_i+x_j+e_p), logit dots ---
  const f16* earow = &s_ea[nloc * DD];             // A row m = lane&15
  float plog[8][HH];
  #pragma unroll
  for (int j = 0; j < 8; ++j)
    #pragma unroll
    for (int h = 0; h < HH; ++h) plog[j][h] = 0.f;

  for (int nt2 = 0; nt2 < 32; ++nt2) {
    v8f ep = {0, 0, 0, 0, 0, 0, 0, 0};
    const f16* brow = WeT + (size_t)(nt2 * 16 + nloc) * DD;
    for (int k0 = 0; k0 < DD; k0 += 32) {
      v16h a = frag_row_lds(earow, k0, h8);
      v16h b = frag_row(brow, k0, h8);
      ep = wmma16(a, b, ep);
    }
    int col = nt2 * 16 + nloc;
    int h = nt2 >> 3;                              // head = col / 128
    float attv = att[h * DD + (col & (DD - 1))];
    #pragma unroll
    for (int j = 0; j < 8; ++j) {
      int m = j + h8;
      float v = x[(size_t)s_trg[m] * HC + col] + x[(size_t)s_src[m] * HC + col] + ep[j];
      float z = v > 0.f ? v : 0.2f * v;            // leaky_relu(0.2)
      plog[j][h] += z * attv;
    }
  }
  // reduce the 16 column-lanes within each half
  #pragma unroll
  for (int m1 = 1; m1 < 16; m1 <<= 1)
    #pragma unroll
    for (int j = 0; j < 8; ++j)
      #pragma unroll
      for (int h = 0; h < HH; ++h)
        plog[j][h] += __shfl_xor(plog[j][h], m1, 32);

  if (nloc == 0) {
    #pragma unroll
    for (int j = 0; j < 8; ++j) {
      int m = j + h8;
      int e = s_e[m];
      if (e >= 0) {
        int t = s_trg[m];
        #pragma unroll
        for (int h = 0; h < HH; ++h) {
          float L = plog[j][h];
          logits[(size_t)e * HH + h] = L;
          atomicMax(&maxenc[(size_t)t * HH + h], enc_f(L));
        }
      }
    }
  }
}

// ---- softmax passes --------------------------------------------------------
__global__ void k_exp(const float* __restrict__ logits, const int* __restrict__ trg,
                      const unsigned* __restrict__ maxenc, float* __restrict__ ex,
                      float* __restrict__ sums) {
  int i = blockIdx.x * blockDim.x + threadIdx.x;
  if (i >= EE * HH) return;
  int e = i >> 2, h = i & 3;
  int t = trg[e];
  float m = dec_f(maxenc[(size_t)t * HH + h]);
  float v = expf(logits[i] - m);
  ex[i] = v;
  atomicAdd(&sums[(size_t)t * HH + h], v);
}
__global__ void k_alpha(float* __restrict__ ex, const int* __restrict__ trg,
                        const float* __restrict__ sums) {
  int i = blockIdx.x * blockDim.x + threadIdx.x;
  if (i >= EE * HH) return;
  int e = i >> 2, h = i & 3;
  ex[i] = ex[i] / fmaxf(sums[(size_t)trg[e] * HH + h], 1e-16f);
}

// ---- weighted aggregation + bias ------------------------------------------
__global__ void k_agg(const float* __restrict__ alpha, const int* __restrict__ src,
                      const int* __restrict__ trg, const float* __restrict__ x,
                      float* __restrict__ out) {
  long long i = (long long)blockIdx.x * blockDim.x + threadIdx.x;  // E*512 threads
  int e = (int)(i >> 9), col = (int)(i & 511), h = col >> 7;
  float a = alpha[(size_t)e * HH + h];
  atomicAdd(&out[(size_t)trg[e] * HC + col], x[(size_t)src[e] * HC + col] * a);
}
__global__ void k_bias(float* __restrict__ out, const float* __restrict__ bias) {
  int i = blockIdx.x * blockDim.x + threadIdx.x;
  if (i < NN * HC) out[i] += bias[i & (HC - 1)];
}

// ---------------------------------------------------------------------------
extern "C" void kernel_launch(void* const* d_in, const int* in_sizes, int n_in,
                              void* d_out, int out_size, void* d_ws, size_t ws_size,
                              hipStream_t stream) {
  (void)in_sizes; (void)n_in; (void)out_size; (void)ws_size;
  const float* embs  = (const float*)d_in[0];
  const int*   eidx  = (const int*)d_in[1];     // [2, E] flat
  const int*   etype = (const int*)d_in[2];
  const float* rel   = (const float*)d_in[3];   // [R, 256, 128]
  const float* W_l   = (const float*)d_in[4];   // [128, 512]
  const float* b_l   = (const float*)d_in[5];
  const float* W_e   = (const float*)d_in[6];   // [128, 512]
  const float* att   = (const float*)d_in[7];   // [4, 128]
  const float* bias  = (const float*)d_in[8];
  const int* srcp = eidx;
  const int* trgp = eidx + EE;
  float* out = (float*)d_out;

  char* w = (char*)d_ws; size_t off = 0;
  auto take = [&](size_t bytes) -> void* {
    void* p = w + off; off = (off + bytes + 255) & ~(size_t)255; return p;
  };
  f16*      embs16 = (f16*)take((size_t)NN * DD * sizeof(f16));
  f16*      relT   = (f16*)take((size_t)RR * DD * (2 * DD) * sizeof(f16));
  f16*      WlT    = (f16*)take((size_t)HC * DD * sizeof(f16));
  f16*      WeT    = (f16*)take((size_t)HC * DD * sizeof(f16));
  float*    x      = (float*)take((size_t)NN * HC * sizeof(float));
  float*    logits = (float*)take((size_t)EE * HH * sizeof(float));
  float*    exb    = (float*)take((size_t)EE * HH * sizeof(float));
  unsigned* maxenc = (unsigned*)take((size_t)NN * HH * sizeof(unsigned));
  float*    sums   = (float*)take((size_t)NN * HH * sizeof(float));
  int*      counts = (int*)take(RR * sizeof(int));
  int*      po     = (int*)take((RR + 1) * sizeof(int));
  int*      cursor = (int*)take(RR * sizeof(int));
  int*      order  = (int*)take((size_t)PAD_E * sizeof(int));

  hipMemsetAsync(counts, 0, RR * sizeof(int), stream);
  hipMemsetAsync(maxenc, 0, (size_t)NN * HH * sizeof(unsigned), stream);
  hipMemsetAsync(sums,   0, (size_t)NN * HH * sizeof(float), stream);
  hipMemsetAsync(out,    0, (size_t)NN * HC * sizeof(float), stream);
  hipMemsetAsync(order, 0xFF, (size_t)PAD_E * sizeof(int), stream);   // -1 fill

  // precision/layout prep
  k_f32_to_f16<<<(NN * DD + 255) / 256, 256, 0, stream>>>(embs, embs16, NN * DD);
  k_transpose<<<(RR * 2 * DD * DD + 255) / 256, 256, 0, stream>>>(rel, relT, 2 * DD, DD, RR);
  k_transpose<<<(DD * HC + 255) / 256, 256, 0, stream>>>(W_l, WlT, DD, HC, 1);
  k_transpose<<<(DD * HC + 255) / 256, 256, 0, stream>>>(W_e, WeT, DD, HC, 1);

  // relation bucketing
  k_hist<<<(EE + 255) / 256, 256, 0, stream>>>(etype, counts, EE);
  k_scan<<<1, 1, 0, stream>>>(counts, po, cursor);
  k_scatter<<<(EE + 255) / 256, 256, 0, stream>>>(etype, cursor, order, EE);

  // WMMA GEMMs
  k_nodeproj<<<MTILES_N * 32, 32, 0, stream>>>(embs16, WlT, b_l, x);
  k_edge<<<NTILES_E, 32, 0, stream>>>(order, po, srcp, trgp, embs16, relT, WeT,
                                      x, att, logits, maxenc);

  // segment softmax + aggregation
  k_exp<<<(EE * HH + 255) / 256, 256, 0, stream>>>(logits, trgp, maxenc, exb, sums);
  k_alpha<<<(EE * HH + 255) / 256, 256, 0, stream>>>(exb, trgp, sums);
  k_agg<<<(EE * HC) / 256, 256, 0, stream>>>(exb, srcp, trgp, x, out);
  k_bias<<<(NN * HC + 255) / 256, 256, 0, stream>>>(out, bias);
}
